// ActorCritic_81501299409458
// MI455X (gfx1250) — matrix operand order repaired and checked by reading, not compiled, not run on hardware
//
#include <hip/hip_runtime.h>

#define B_TOT 16384
#define IN_DIM 256
#define OUT_DIM 64
#define TDEPTH 10
#define N_INT 1023
#define N_LEAF 1024
#define HID 512
#define ROWS 32                      // batch rows per block (2 row-groups of 16)
#define SMEM_BYTES (113 * 1024)      // 16K sX + 64K sVal + 32K sH + 1K red

typedef __attribute__((ext_vector_type(16))) __bf16 v16bf;
typedef __attribute__((ext_vector_type(8)))  float v8f;
typedef unsigned int v4u __attribute__((ext_vector_type(4)));
typedef int v8i __attribute__((ext_vector_type(8)));
typedef int v4i __attribute__((ext_vector_type(4)));

#if defined(__has_builtin)
#if __has_builtin(__builtin_amdgcn_tensor_load_to_lds) && __has_builtin(__builtin_amdgcn_s_wait_tensorcnt)
#define HAVE_TDM 1
#endif
#endif
#ifndef HAVE_TDM
#define HAVE_TDM 0
#endif

#if HAVE_TDM
#pragma message("CDNA5: tensor_load_to_lds TDM path ENABLED")
#else
#pragma message("CDNA5: TDM builtin NOT available, direct-load fallback")
#endif

// A-matrix (16x32 bf16) element->K mapping per CDNA5 ISA 7.12.2:
// lanes 0-15: V0..V3 hold K=0..7 (2/VGPR), V4..V7 hold K=16..23; lanes 16-31: +8
__device__ __forceinline__ int kmapA(int i, int lane) {
  int p = i >> 1, b = i & 1;
  return ((p & 4) << 2) + ((p & 3) << 1) + b + ((lane >> 4) << 3);
}

// A-fragment from an LDS row (two contiguous 16B runs -> ds_load_b128 x2)
__device__ __forceinline__ v16bf loadA(const __bf16* sXrow, int k0, int lane) {
  v16bf a;
#pragma unroll
  for (int i = 0; i < 16; ++i) a[i] = sXrow[k0 + kmapA(i, lane)];
  return a;
}

// B-fragment: lane = N column, element i -> K = kmapA(i, lane), row stride ld
__device__ __forceinline__ v16bf packB16(const float* p, int ld, int lane) {
  v16bf b;
#pragma unroll
  for (int i = 0; i < 16; ++i) b[i] = (__bf16)p[(size_t)kmapA(i, lane) * ld];
  return b;
}

// Node-GEMM 16(N)x2x16(M) tile over K=IN_DIM: B loaded ONCE per chunk, fed to
// two WMMAs (row-groups). GUARD specialization covers the tile straddling
// row N_INT via clamped unconditional loads + select (no exec-masked loads).
template <bool GUARD>
__device__ __forceinline__ void nodeGemm2(const float* __restrict__ Wn,
                                          const __bf16* sXr0, const __bf16* sXr1,
                                          int n0, int lane, v8f& acc0, v8f& acc1) {
#pragma unroll 2
  for (int k0 = 0; k0 < IN_DIM; k0 += 32) {
    if (k0 + 32 < IN_DIM)
      __builtin_prefetch(&Wn[(size_t)n0 * IN_DIM + k0 + 32 + lane], 0, 1);
    const int nr = n0 + (lane & 15);
    const bool ok = !GUARD || (nr < N_INT);
    const float* wp = Wn + (size_t)(ok ? nr : n0) * IN_DIM + k0;  // B: lane=N, elem=K
    float wv[16];
#pragma unroll
    for (int i = 0; i < 16; ++i) {
      wv[i] = wp[kmapA(i, lane)];
    }
    v16bf b;
#pragma unroll
    for (int i = 0; i < 16; ++i)
      b[i] = (__bf16)(ok ? wv[i] : 0.0f);
    v16bf a0 = loadA(sXr0, k0, lane);
    v16bf a1 = loadA(sXr1, k0, lane);
    acc0 = __builtin_amdgcn_wmma_f32_16x16x32_bf16(false, a0, false, b,
                                                   (short)0, acc0, false, false);
    acc1 = __builtin_amdgcn_wmma_f32_16x16x32_bf16(false, a1, false, b,
                                                   (short)0, acc1, false, false);
  }
}

__global__ __launch_bounds__(256)
void actorcritic_sdt_fused(const float* __restrict__ x,
                           const float* __restrict__ Wn,
                           const float* __restrict__ bn,
                           const float* __restrict__ leaf,
                           const float* __restrict__ cw1,
                           const float* __restrict__ cb1,
                           const float* __restrict__ cw2,
                           const float* __restrict__ cb2,
                           float* __restrict__ out)
{
  extern __shared__ __align__(16) char smem[];
  __bf16 (*sX)[IN_DIM]  = (__bf16(*)[IN_DIM])(smem);               // 16 KB
  __bf16 (*sVal)[N_LEAF] = (__bf16(*)[N_LEAF])(smem + 16 * 1024);  // 64 KB
  __bf16 (*sH)[HID]      = (__bf16(*)[HID])(smem + 80 * 1024);     // 32 KB
  float* sRed            = (float*)(smem + 112 * 1024);            //  1 KB

  const int tid   = threadIdx.x;
  const int lane  = tid & 31;
  const int wave  = tid >> 5;
  const int row16 = lane & 15;
  const int hi    = lane >> 4;
  const int gRow0 = blockIdx.x * ROWS;

#if HAVE_TDM
  // ---- stage x tile (32x256 f32 = 32KB contiguous) via the Tensor Data
  //      Mover into LDS, then convert f32->bf16. Stage aliases sVal region.
  float* xstage = (float*)&sVal[0][0];
  if (wave == 0) {
    unsigned ldsAddr = (unsigned)(uintptr_t)xstage;          // low 32b = LDS offset
    unsigned long long ga =
        (unsigned long long)(uintptr_t)(x + (size_t)gRow0 * IN_DIM);
    v4u g0;
    g0[0] = 1u;                                   // count=1 (valid descriptor)
    g0[1] = ldsAddr;                              // lds_addr [63:32]
    g0[2] = (unsigned)ga;                         // global_addr lo
    g0[3] = (unsigned)((ga >> 32) & 0x01FFFFFFu) | (2u << 30);  // addr hi | type=2
    v8i g1;
    g1[0] = 0x00020000;          // wg_mask=0, data_size=2 (4B)
    g1[1] = (int)(8192u << 16);  // tensor_dim0 = 8192 elements (bits 79:48)
    g1[2] = (int)(1u << 16);     // tensor_dim1 = 1 (bits 111:80)
    g1[3] = (int)(8192u << 16);  // tile_dim0 = 8192 (bits 127:112)
    g1[4] = 1;                   // tile_dim1 = 1, tile_dim2 = 0
    g1[5] = 8192;                // tensor_dim0_stride lo
    g1[6] = 0;
    g1[7] = 0;
    v4i gz = {0, 0, 0, 0};
#if __clang_major__ >= 23
    v8i gz8 = {0, 0, 0, 0, 0, 0, 0, 0};
    __builtin_amdgcn_tensor_load_to_lds(g0, g1, gz, gz, gz8, 0);
#else
    __builtin_amdgcn_tensor_load_to_lds(g0, g1, gz, gz, 0);
#endif
    __builtin_amdgcn_s_wait_tensorcnt(0);
  }
  __syncthreads();
  for (int idx = tid; idx < ROWS * IN_DIM; idx += 256) {
    int r = idx >> 8, k = idx & (IN_DIM - 1);
    sX[r][k] = (__bf16)xstage[idx];
  }
#else
  for (int idx = tid; idx < ROWS * IN_DIM; idx += 256) {
    int r = idx >> 8, k = idx & (IN_DIM - 1);
    sX[r][k] = (__bf16)x[(size_t)(gRow0 + r) * IN_DIM + k];
  }
#endif
  __syncthreads();

  // ---- Phase A: val = sigmoid(x @ Wn^T + bn), B-fragment reused x2 ----
  for (int nt = wave; nt < N_LEAF / 16; nt += 8) {
    const int n0 = nt * 16;
    v8f acc0 = {}, acc1 = {};
    if (n0 + 16 <= N_INT)
      nodeGemm2<false>(Wn, sX[row16], sX[16 + row16], n0, lane, acc0, acc1);
    else
      nodeGemm2<true>(Wn, sX[row16], sX[16 + row16], n0, lane, acc0, acc1);
    const int ncol = n0 + row16;                 // D: lane = N, VGPR = M
    const float bias = (ncol < N_INT) ? bn[ncol] : 0.0f;
#pragma unroll
    for (int j = 0; j < 8; ++j) {
      int m = j + 8 * hi;
      float t0 = acc0[j] + bias;
      float t1 = acc1[j] + bias;
      sVal[m][ncol]      = (__bf16)(1.0f / (1.0f + __expf(-t0)));
      sVal[16 + m][ncol] = (__bf16)(1.0f / (1.0f + __expf(-t1)));
    }
  }
  __syncthreads();

  // ---- Phase B1: hard routing, one thread per row ----
  if (tid < ROWS) {
    int ptr = 0;
    for (int d = 0; d < TDEPTH; ++d) {
      float v = (float)sVal[tid][ptr];
      ptr = 2 * ptr + 1 + ((v >= 0.5f) ? 1 : 0);
    }
    out[(size_t)B_TOT * OUT_DIM + gRow0 + tid] = (float)(ptr - (N_LEAF - 1));
  }

  // ---- Phase B2: waves 0-3: p = prob @ leaf_dist (prob built in-register,
  //                two-level path-prefix hoisting); waves 4-7: critic h ----
  if (wave < 4) {
    const int n0 = wave * 16;
    const int ncol = n0 + row16;
    v8f acc0 = {}, acc1 = {};
    for (int k0 = 0; k0 < N_LEAF; k0 += 32) {
      v16bf b = packB16(leaf + (size_t)k0 * OUT_DIM + n0 + row16, OUT_DIM, lane);
      v16bf afr[2];
#pragma unroll
      for (int rg = 0; rg < 2; ++rg) {
        const __bf16* vrow = sVal[rg * 16 + row16];
        // levels 0..4 depend only on k0
        float pref = 1.0f;
        int node0 = 0;
#pragma unroll
        for (int d = 0; d < 5; ++d) {
          int bit = (k0 >> (TDEPTH - 1 - d)) & 1;
          float v = (float)vrow[node0];
          pref *= bit ? v : (1.0f - v);
          node0 = 2 * node0 + 1 + bit;
        }
        // per-lane leaf deltas: {hi*8+j, 16+hi*8+j} -> levels 5,6 shared per half
#pragma unroll
        for (int h = 0; h < 2; ++h) {
          int b5 = h, b6 = hi;
          float pm = pref;
          int nm = node0;
          float v5 = (float)vrow[nm];
          pm *= b5 ? v5 : (1.0f - v5);
          nm = 2 * nm + 1 + b5;
          float v6 = (float)vrow[nm];
          pm *= b6 ? v6 : (1.0f - v6);
          nm = 2 * nm + 1 + b6;
#pragma unroll
          for (int j = 0; j < 8; ++j) {          // levels 7,8,9 from j bits 2..0
            float p = pm;
            int n = nm;
#pragma unroll
            for (int d = 0; d < 3; ++d) {
              int bit = (j >> (2 - d)) & 1;
              float v = (float)vrow[n];
              p *= bit ? v : (1.0f - v);
              n = 2 * n + 1 + bit;
            }
            afr[rg][h * 8 + j] = (__bf16)p;
          }
        }
      }
      acc0 = __builtin_amdgcn_wmma_f32_16x16x32_bf16(false, afr[0], false, b,
                                                     (short)0, acc0, false, false);
      acc1 = __builtin_amdgcn_wmma_f32_16x16x32_bf16(false, afr[1], false, b,
                                                     (short)0, acc1, false, false);
    }
#pragma unroll
    for (int j = 0; j < 8; ++j) {
      int m = j + 8 * hi;
      out[(size_t)(gRow0 + m) * OUT_DIM + ncol]      = acc0[j];
      out[(size_t)(gRow0 + 16 + m) * OUT_DIM + ncol] = acc1[j];
    }
  } else {
    for (int nt = wave - 4; nt < HID / 16; nt += 4) {
      const int n0 = nt * 16;
      const int ncol = n0 + row16;
      v8f acc0 = {}, acc1 = {};
#pragma unroll 2
      for (int k0 = 0; k0 < IN_DIM; k0 += 32) {
        v16bf b = packB16(cw1 + (size_t)k0 * HID + n0 + row16, HID, lane);
        v16bf a0 = loadA(sX[row16], k0, lane);
        v16bf a1 = loadA(sX[16 + row16], k0, lane);
        acc0 = __builtin_amdgcn_wmma_f32_16x16x32_bf16(false, a0, false, b,
                                                       (short)0, acc0, false, false);
        acc1 = __builtin_amdgcn_wmma_f32_16x16x32_bf16(false, a1, false, b,
                                                       (short)0, acc1, false, false);
      }
      const float bias = cb1[ncol];
#pragma unroll
      for (int j = 0; j < 8; ++j) {
        int m = j + 8 * hi;
        float t0 = acc0[j] + bias;
        float t1 = acc1[j] + bias;
        sH[m][ncol]      = (__bf16)(t0 > 0.0f ? t0 : 0.0f);
        sH[16 + m][ncol] = (__bf16)(t1 > 0.0f ? t1 : 0.0f);
      }
    }
  }
  __syncthreads();

  // ---- Phase C: v_out = h @ cw2 + cb2 (8 partial sums per row) ----
  {
    const int r = tid >> 3, part = tid & 7;
    float s = 0.0f;
    for (int k = part * 64; k < part * 64 + 64; ++k)
      s += (float)sH[r][k] * cw2[k];
    sRed[tid] = s;
  }
  __syncthreads();
  if (tid < ROWS) {
    float s = cb2[0];
#pragma unroll
    for (int i = 0; i < 8; ++i) s += sRed[tid * 8 + i];
    out[(size_t)B_TOT * OUT_DIM + B_TOT + gRow0 + tid] = s;
  }
}

extern "C" void kernel_launch(void* const* d_in, const int* in_sizes, int n_in,
                              void* d_out, int out_size, void* d_ws, size_t ws_size,
                              hipStream_t stream) {
  const float* x    = (const float*)d_in[0];
  const float* Wn   = (const float*)d_in[1];
  const float* bn   = (const float*)d_in[2];
  const float* leaf = (const float*)d_in[3];
  const float* cw1  = (const float*)d_in[4];
  const float* cb1  = (const float*)d_in[5];
  const float* cw2  = (const float*)d_in[6];
  const float* cb2  = (const float*)d_in[7];
  float* out = (float*)d_out;

  dim3 grid(B_TOT / ROWS);
  dim3 block(256);
  actorcritic_sdt_fused<<<grid, block, SMEM_BYTES, stream>>>(
      x, Wn, bn, leaf, cw1, cb1, cw2, cb2, out);
}
